// MultiviewCrossAttnFeature_54254026883691
// MI455X (gfx1250) — matrix operand order repaired and checked by reading, not compiled, not run on hardware
//
#include <hip/hip_runtime.h>

// ---------------- problem constants ----------------
#define N_VOXC   50000
#define M_VIEW   12
#define F_DIM    256
#define B_TOT    16384
#define NHEAD    8
#define DKV      64
#define PE_ORD   8
#define KAUG     272          // F_DIM + 16
#define KPAD     288          // KAUG padded to 9*32 (zero rows 272..287)
#define TB       4            // batch elements per workgroup
#define ROWS     48           // TB * M_VIEW  (== 3 row-tiles of 16)
#define NTHREADS 256          // 8 waves (wave32) -> one head per wave

// ---------------- workspace layout (d_ws) ----------------
#define WS_MM    0
#define WS_WKT   1024         // u16 WkT[512][288], rows>=272 zero
#define WS_WVT   295936       // u16 WvT[512][256]

// ---------------- LDS layout (dynamic) ----------------
#define SMEM_Q      0        //  8192 B : q        TB*512 f32
#define SMEM_PE     8192     //   384 B : pe       TB*24  f32
#define SMEM_MASK   8576     //   192 B : mask     ROWS   f32
#define SMEM_LOGITS 8768     //  1536 B : logits   ROWS*8 f32
#define SMEM_P      10304    //  1536 B : p        TB*8*12 f32
#define SMEM_IDX    11840    //    32 B : idx      TB int
#define SMEM_AFEAT  11872    // 25344 B : Afeat    ROWS*264 u16
#define SMEM_AEXT   37216    //  3840 B : Aext     ROWS*40  u16
#define SMEM_WSTG   41056    // 81920 B : 8 waves * 2 bufs * (64*40) u16
#define SMEM_BYTES  122976

#define ASTRIDE 264   // bf16 per Afeat row (256 + 8 pad), 528 B
#define ESTRIDE 40    // bf16 per Aext row (32 + 8 pad),   80 B
#define WSTRIDE 40    // bf16 per staged-W column (32 + 8 pad), 80 B == TDM pad layout
#define WBUF_U16 2560 // u16 per W buffer (64*40)

typedef __attribute__((ext_vector_type(16))) __bf16 v16bf;
typedef __attribute__((ext_vector_type(8)))  float  v8f;
typedef __attribute__((ext_vector_type(4)))  unsigned v4u;
typedef __attribute__((ext_vector_type(8)))  int      v8i;
typedef __attribute__((ext_vector_type(4)))  int      v4i;

// f32 -> bf16 (round-to-nearest-even), stored as u16
__device__ __forceinline__ unsigned short f2bf(float f) {
    unsigned u = __float_as_uint(f);
    u += 0x7FFFu + ((u >> 16) & 1u);
    return (unsigned short)(u >> 16);
}

// assemble a v16bf fragment from two 16-byte LDS chunks
__device__ __forceinline__ v16bf load_frag2(const unsigned short* p0,
                                            const unsigned short* p1) {
    union { uint4 u[2]; v16bf v; } f;
    f.u[0] = *(const uint4*)p0;
    f.u[1] = *(const uint4*)p1;
    return f.v;
}

__device__ __forceinline__ void zero_acc(v8f (&acc)[3][4]) {
#pragma unroll
    for (int rt = 0; rt < 3; ++rt)
#pragma unroll
        for (int nt = 0; nt < 4; ++nt)
#pragma unroll
            for (int j = 0; j < 8; ++j) acc[rt][nt][j] = 0.f;
}

// ---- Tensor Data Mover: load one 32(K) x 64(N) bf16 tile from a [n][k]
// tensor (row pitch strideElems) into LDS at ldsOff, padding each 32-element
// row with 4 DWORDs so the LDS column pitch is WSTRIDE (40 halfwords).
__device__ __forceinline__ void tdm_load_tile(unsigned ldsOff,
                                              const unsigned short* gsrc,
                                              unsigned strideElems) {
    unsigned long long ga = (unsigned long long)(uintptr_t)gsrc;
    unsigned lo = __builtin_amdgcn_readfirstlane((unsigned)ga);
    unsigned hi = __builtin_amdgcn_readfirstlane((unsigned)(ga >> 32));
    ldsOff      = __builtin_amdgcn_readfirstlane(ldsOff);

    v4u g0;
    g0[0] = 1u;                                   // count=1 (valid user D#)
    g0[1] = ldsOff;                               // lds_addr (bytes)
    g0[2] = lo;                                   // global_addr[31:0]
    g0[3] = (hi & 0x01FFFFFFu) | (2u << 30);      // global_addr[56:32] | type=2

    v8i g1;
    g1[0] = (int)((1u << 16)      // data_size = 2B
                | (1u << 20)      // pad_enable
                | (3u << 22)      // pad_interval: 16 DWORDs (one 32-elem row)
                | (3u << 25));    // pad_amount:   4 DWORDs (8 halfwords)
    g1[1] = (int)(32u << 16);     // tensor_dim0 = 32 (tile always in-bounds)
    g1[2] = (int)(64u << 16);     // tensor_dim1 = 64
    g1[3] = (int)(32u << 16);     // tile_dim0 = 32 (K chunk)
    g1[4] = 64;                   // tile_dim1 = 64 (N columns)
    g1[5] = (int)strideElems;     // tensor_dim0_stride (elements)
    g1[6] = 0;
    g1[7] = 0;

    v4i gz4 = {0, 0, 0, 0};
    v8i gz8 = {0, 0, 0, 0, 0, 0, 0, 0};
    __builtin_amdgcn_tensor_load_to_lds(g0, g1, gz4, gz4, gz8, 0);
}

// one K-step (depth 32) of the 48x64 tile: 3 row-tiles x 4 n-tiles of WMMA.
// A layout (16-bit, 16x32): lane M = lane%16; lanes<16 K{0..7,16..23}, lanes>=16 K{8..15,24..31}
// B layout (16-bit, 32x16): lane N = lane%16; lanes<16 K{0..7,16..23}, lanes>=16 K{8..15,24..31}
__device__ __forceinline__ void wmma_step(v8f (&acc)[3][4],
                                          const unsigned short* wt,   // LDS, this buffer
                                          const unsigned short* Abase, int aStride,
                                          int aColOff, int n, int kOffA, int kOffB) {
    v16bf bf[4];
#pragma unroll
    for (int nt = 0; nt < 4; ++nt) {
        const unsigned short* bp = wt + (nt * 16 + n) * WSTRIDE + kOffB;
        bf[nt] = load_frag2(bp, bp + 16);
    }
#pragma unroll
    for (int rt = 0; rt < 3; ++rt) {
        const unsigned short* ap = Abase + (rt * 16 + n) * aStride + aColOff + kOffA;
        v16bf av = load_frag2(ap, ap + 16);
#pragma unroll
        for (int nt = 0; nt < 4; ++nt)
            acc[rt][nt] = __builtin_amdgcn_wmma_f32_16x16x32_bf16(
                false, av, false, bf[nt], (short)0, acc[rt][nt], false, false);
    }
}

// logits: dot K[row, h*64+0..63] with q; templated on HALF so register
// indexing stays static. Shuffle partners stay inside the 16-lane half.
template <int HALF>
__device__ __forceinline__ void logits_pass(const v8f (&acc)[3][4],
                                            const float (&qreg)[TB][4],
                                            const float* maskLds, float* logitsLds,
                                            int h, int n) {
#pragma unroll
    for (int rt = 0; rt < 3; ++rt) {
#pragma unroll
        for (int j = 0; j < 8; ++j) {
            const int row = rt * 16 + j + 8 * HALF;
            const int b = row / M_VIEW;
            float val = acc[rt][0][j] * qreg[b][0] + acc[rt][1][j] * qreg[b][1]
                      + acc[rt][2][j] * qreg[b][2] + acc[rt][3][j] * qreg[b][3];
            val += __shfl_xor(val, 1, 32);
            val += __shfl_xor(val, 2, 32);
            val += __shfl_xor(val, 4, 32);
            val += __shfl_xor(val, 8, 32);
            if (n == 0)
                logitsLds[row * NHEAD + h] = val * 0.125f + maskLds[row];
        }
    }
}

// out[b, h*64+d] += sum_m p[b,h,m] * V[row(b,m), d]
template <int HALF>
__device__ __forceinline__ void out_reduce(const v8f (&acc)[3][4], const float* pLds,
                                           int h, float (&sO)[TB][4]) {
#pragma unroll
    for (int rt = 0; rt < 3; ++rt) {
#pragma unroll
        for (int j = 0; j < 8; ++j) {
            const int row = rt * 16 + j + 8 * HALF;
            const int b = row / M_VIEW;
            const int m = row % M_VIEW;
            float pv = pLds[(b * NHEAD + h) * M_VIEW + m];
#pragma unroll
            for (int nt = 0; nt < 4; ++nt) sO[b][nt] += pv * acc[rt][nt][j];
        }
    }
}

// ---------------- kernel 0: voxel min/max ----------------
__global__ void voxel_minmax_kernel(const float* __restrict__ voxels,
                                    float* __restrict__ mm, int n) {
    __shared__ float red[1024];
    int tid = threadIdx.x;
    for (int c = 0; c < 3; ++c) {
        float mn = 3.0e38f, mx = -3.0e38f;
        for (int i = tid; i < n; i += 1024) {
            float v = voxels[(size_t)i * 3 + c];
            mn = fminf(mn, v);
            mx = fmaxf(mx, v);
        }
        red[tid] = mn; __syncthreads();
        for (int s = 512; s > 0; s >>= 1) {
            if (tid < s) red[tid] = fminf(red[tid], red[tid + s]);
            __syncthreads();
        }
        if (tid == 0) mm[c] = red[0];
        __syncthreads();
        red[tid] = mx; __syncthreads();
        for (int s = 512; s > 0; s >>= 1) {
            if (tid < s) red[tid] = fmaxf(red[tid], red[tid + s]);
            __syncthreads();
        }
        if (tid == 0) mm[3 + c] = red[0];
        __syncthreads();
    }
}

// ---------------- kernel 1: weight pre-transpose to bf16 [n][k] ----------------
__global__ void prep_weights_kernel(const float* __restrict__ Wk,
                                    const float* __restrict__ Wv,
                                    unsigned short* __restrict__ WkT,
                                    unsigned short* __restrict__ WvT) {
    int id = blockIdx.x * blockDim.x + threadIdx.x;
    if (id < 512 * KPAD) {
        int nIdx = id / KPAD, k = id % KPAD;
        float v = (k < KAUG) ? Wk[(size_t)k * 512 + nIdx] : 0.f;
        WkT[id] = f2bf(v);
    }
    if (id < 512 * F_DIM) {
        int nIdx = id >> 8, k = id & 255;
        WvT[id] = f2bf(Wv[(size_t)k * 512 + nIdx]);
    }
}

// ---------------- kernel 2: fused multiview cross attention ----------------
__global__ void __launch_bounds__(NTHREADS)
mv_xattn_kernel(const int* __restrict__ vox_indices,
                const float* __restrict__ voxels,
                const float* __restrict__ vox_feat,
                const float* __restrict__ scores,
                const int* __restrict__ cam_ids,
                const float* __restrict__ extr,
                const float* __restrict__ Wq,
                const unsigned short* __restrict__ WkT,
                const unsigned short* __restrict__ WvT,
                const float* __restrict__ mm,
                float* __restrict__ out) {
    extern __shared__ char smem[];
    float*          qLds      = (float*)(smem + SMEM_Q);
    float*          peLds     = (float*)(smem + SMEM_PE);
    float*          maskLds   = (float*)(smem + SMEM_MASK);
    float*          logitsLds = (float*)(smem + SMEM_LOGITS);
    float*          pLds      = (float*)(smem + SMEM_P);
    int*            idxLds    = (int*)(smem + SMEM_IDX);
    unsigned short* Afeat     = (unsigned short*)(smem + SMEM_AFEAT);
    unsigned short* Aext      = (unsigned short*)(smem + SMEM_AEXT);
    unsigned short* Wstg      = (unsigned short*)(smem + SMEM_WSTG);

    const int tid = threadIdx.x;
    const int b0 = blockIdx.x * TB;

    if (tid < TB) idxLds[tid] = vox_indices[b0 + tid];
    __syncthreads();

    // ---- phase 0a: gather features, convert to bf16 into LDS ----
    for (int e = tid; e < ROWS * (F_DIM / 4); e += NTHREADS) {
        int row = e >> 6;
        int k = (e & 63) * 4;
        int b = row / M_VIEW, m = row % M_VIEW;
        const float4 f4 = *(const float4*)(vox_feat +
                              ((size_t)idxLds[b] * M_VIEW + m) * F_DIM + k);
        unsigned lo = (unsigned)f2bf(f4.x) | ((unsigned)f2bf(f4.y) << 16);
        unsigned hi = (unsigned)f2bf(f4.z) | ((unsigned)f2bf(f4.w) << 16);
        unsigned* dst = (unsigned*)(Afeat + row * ASTRIDE + k);
        dst[0] = lo;
        dst[1] = hi;
    }

    // ---- phase 0b: extrinsics rows (K 256..271) + zero pad + mask ----
    if (tid < ROWS) {
        int row = tid, b = row / M_VIEW, m = row % M_VIEW;
        int cam = cam_ids[idxLds[b] * M_VIEW + m];
        const float* ep = extr + (size_t)cam * 16;
        unsigned* dst = (unsigned*)(Aext + row * ESTRIDE);
#pragma unroll
        for (int j = 0; j < 16; j += 2)
            dst[j >> 1] = (unsigned)f2bf(ep[j]) | ((unsigned)f2bf(ep[j + 1]) << 16);
#pragma unroll
        for (int j = 8; j < 16; ++j) dst[j] = 0;
        float s = scores[idxLds[b] * M_VIEW + m];
        maskLds[row] = (s < 0.f) ? -__builtin_inff() : 0.f;
    }

    // ---- phase 0c: positional encoding ----
    if (tid < TB * 3) {
        int b = tid / 3, c = tid % 3;
        float v = voxels[(size_t)idxLds[b] * 3 + c];
        float x = (v - mm[c]) / (mm[3 + c] - mm[c]) - 0.5f;
        float f = 1.0f;
#pragma unroll
        for (int p = 0; p < PE_ORD; ++p) {
            peLds[b * 24 + p * 3 + c] = __sinf(x * f);
            f *= 2.f;
        }
    }
    __syncthreads();

    // ---- phase 0d: q = pe @ Wq  (TB x 512) ----
    for (int e = tid; e < TB * 512; e += NTHREADS) {
        int b = e >> 9, col = e & 511;
        float sum = 0.f;
#pragma unroll
        for (int f = 0; f < 24; ++f) sum += peLds[b * 24 + f] * Wq[f * 512 + col];
        qLds[b * 512 + col] = sum;
    }
    __syncthreads();

    // ---- per-wave setup: wave w owns head h = w ----
    const int wave = tid >> 5, lane = tid & 31;
    const int h = wave;
    const int half = lane >> 4, n = lane & 15;
    const int kOffA = 8 * half;    // A fragment K base within step
    const int kOffB = 8 * half;    // B fragment K base within step

    // single LDS-provenance base pointer; buffers selected via integer offset
    unsigned short* wavebase = Wstg + wave * 2 * WBUF_U16;
    const unsigned wtOff0 = (unsigned)(uintptr_t)wavebase;
    const unsigned wtOff1 = wtOff0 + WBUF_U16 * 2;   // bytes
    const unsigned short* WkH = WkT + (size_t)(h * DKV) * KPAD;   // head slice [64][288]
    const unsigned short* WvH = WvT + (size_t)(h * DKV) * F_DIM;  // head slice [64][256]

    float qreg[TB][4];
#pragma unroll
    for (int b = 0; b < TB; ++b)
#pragma unroll
        for (int nt = 0; nt < 4; ++nt)
            qreg[b][nt] = qLds[b * 512 + h * DKV + nt * 16 + n];

    v8f acc[3][4];
    zero_acc(acc);

    // ---- pass 1: K-tile GEMM (9 K-steps over WkT) + logits ----
    tdm_load_tile(wtOff0, WkH, KPAD);                       // prefetch ks=0
    for (int ks = 0; ks < 8; ++ks) {
        int nxt = ks + 1;
        asm volatile("s_wait_dscnt 0x0" ::: "memory");      // reads of next buf done
        tdm_load_tile((nxt & 1) ? wtOff1 : wtOff0, WkH + nxt * 32, KPAD);
        __builtin_amdgcn_s_wait_tensorcnt(1);               // current buf ready
        wmma_step(acc, wavebase + ((ks & 1) ? WBUF_U16 : 0),
                  Afeat, ASTRIDE, ks * 32, n, kOffA, kOffB);
    }
    __builtin_amdgcn_s_wait_tensorcnt(0);                   // ext-step tile ready
    wmma_step(acc, wavebase, Aext, ESTRIDE, 0, n, kOffA, kOffB);

    // prefetch pass-2 first V tile so the DMA hides under logits/softmax
    asm volatile("s_wait_dscnt 0x0" ::: "memory");
    tdm_load_tile(wtOff0, WvH, F_DIM);

    if (half == 0) logits_pass<0>(acc, qreg, maskLds, logitsLds, h, n);
    else           logits_pass<1>(acc, qreg, maskLds, logitsLds, h, n);
    __syncthreads();

    // ---- masked softmax over M=12 (one thread per (b, h)) ----
    if (tid < TB * NHEAD) {
        int b = tid >> 3, hh = tid & 7;
        float l[M_VIEW], mx = -__builtin_inff();
#pragma unroll
        for (int m = 0; m < M_VIEW; ++m) {
            l[m] = logitsLds[(b * M_VIEW + m) * NHEAD + hh];
            mx = fmaxf(mx, l[m]);
        }
        float ssum = 0.f;
#pragma unroll
        for (int m = 0; m < M_VIEW; ++m) {
            l[m] = __expf(l[m] - mx);
            ssum += l[m];
        }
        float inv = 1.f / ssum;
#pragma unroll
        for (int m = 0; m < M_VIEW; ++m)
            pLds[(b * NHEAD + hh) * M_VIEW + m] = l[m] * inv;
    }
    __syncthreads();

    // ---- pass 2: V-tile GEMM (8 K-steps over WvT) + p-weighted reduction ----
    zero_acc(acc);
    for (int ks = 0; ks < 8; ++ks) {
        if (ks < 7) {
            int nxt = ks + 1;
            asm volatile("s_wait_dscnt 0x0" ::: "memory");
            tdm_load_tile((nxt & 1) ? wtOff1 : wtOff0, WvH + nxt * 32, F_DIM);
            __builtin_amdgcn_s_wait_tensorcnt(1);
        } else {
            __builtin_amdgcn_s_wait_tensorcnt(0);
        }
        wmma_step(acc, wavebase + ((ks & 1) ? WBUF_U16 : 0),
                  Afeat, ASTRIDE, ks * 32, n, kOffA, kOffB);
    }

    float sO[TB][4];
#pragma unroll
    for (int b = 0; b < TB; ++b)
#pragma unroll
        for (int nt = 0; nt < 4; ++nt) sO[b][nt] = 0.f;

    if (half == 0) out_reduce<0>(acc, pLds, h, sO);
    else           out_reduce<1>(acc, pLds, h, sO);

#pragma unroll
    for (int b = 0; b < TB; ++b) {
#pragma unroll
        for (int nt = 0; nt < 4; ++nt) {
            float v = sO[b][nt];
            v += __shfl_xor(v, 16, 32);   // combine the two row-halves
            if (half == 0)
                out[(size_t)(b0 + b) * (NHEAD * DKV) + h * DKV + nt * 16 + n] = v;
        }
    }
}

// ---------------- host entry ----------------
extern "C" void kernel_launch(void* const* d_in, const int* in_sizes, int n_in,
                              void* d_out, int out_size, void* d_ws, size_t ws_size,
                              hipStream_t stream) {
    const int*   vox_indices = (const int*)d_in[0];
    const float* voxels      = (const float*)d_in[1];
    const float* vox_feat    = (const float*)d_in[2];
    const float* scores      = (const float*)d_in[3];
    const int*   cam_ids     = (const int*)d_in[4];
    const float* extr        = (const float*)d_in[5];
    const float* Wq          = (const float*)d_in[6];
    const float* Wk          = (const float*)d_in[7];
    const float* Wv          = (const float*)d_in[8];
    float*       outp        = (float*)d_out;

    float*          mm  = (float*)((char*)d_ws + WS_MM);
    unsigned short* WkT = (unsigned short*)((char*)d_ws + WS_WKT);
    unsigned short* WvT = (unsigned short*)((char*)d_ws + WS_WVT);

    voxel_minmax_kernel<<<1, 1024, 0, stream>>>(voxels, mm, N_VOXC);
    prep_weights_kernel<<<(512 * KPAD + 255) / 256, 256, 0, stream>>>(Wk, Wv, WkT, WvT);
    mv_xattn_kernel<<<B_TOT / TB, NTHREADS, SMEM_BYTES, stream>>>(
        vox_indices, voxels, vox_feat, scores, cam_ids, extr, Wq, WkT, WvT, mm, outp);
}
